// Transformer_NMT_MH_76682346103551
// MI455X (gfx1250) — matrix operand order repaired and checked by reading, not compiled, not run on hardware
//
#include <hip/hip_runtime.h>
#include <math.h>

// ---------------------------------------------------------------------------
// Transformer NMT forward on gfx1250 (MI455X), bf16 WMMA everywhere.
//
// Roofline: ~550 GFLOP, ~0.9GB of fp32 weights each touched once ->
// stream f32 from HBM (float4 loads), convert to bf16 in LDS, accumulate
// f32 via v_wmma_f32_16x16x32_bf16 (wave32).  GEMM block tile 64x128:
// 4 WMMAs per wave per K-step against one held B fragment.
//
// Input flattening assumption (setup_inputs() recursive dict order):
//   0: src_indices [B*S]   1: tgt_indices [B*S]
//   2: src_emb  3: tgt_emb  4: out.W  5: out.b
//   6..69:   enc[0..3], each 16 arrays:
//            q.W q.b k.W k.b v.W v.b o.W o.b ff1.W ff1.b ff2.W ff2.b
//            ln1_g ln1_b ln2_g ln2_b
//   70..197: dec[0..3], each dd(16) then ed(16) in the same order.
// ---------------------------------------------------------------------------

#define DMODEL 1024
#define DINT   1024
#define NHEAD  16
#define DHEAD  64
#define NLAYER 4
#define VOCAB  32000
#define BATCH  4
#define SEQ    512
#define NTOK   (BATCH * SEQ)      // 2048 tokens
#define NCHUNK (VOCAB / 128)      // 250 vocab chunks of 128

typedef __attribute__((ext_vector_type(16))) __bf16 v16bf;
typedef __attribute__((ext_vector_type(8)))  float  v8f;

// ---- bf16 helpers (RNE via integer math)
static __device__ __forceinline__ unsigned short f2bfu(float f) {
  unsigned u = __builtin_bit_cast(unsigned, f);
  unsigned r = (u + 0x7FFFu + ((u >> 16) & 1u)) >> 16;
  return (unsigned short)r;
}
static __device__ __forceinline__ unsigned pack2bf(float a, float b) {
  return (unsigned)f2bfu(a) | ((unsigned)f2bfu(b) << 16);
}
static __device__ __forceinline__ __bf16 u2bf(unsigned short h) {
  return __builtin_bit_cast(__bf16, h);
}
static __device__ __forceinline__ __bf16 fbf(float f) { return u2bf(f2bfu(f)); }

// ---- 16-lane butterfly reductions (wave32: lanes 0-15 / 16-31 independent)
static __device__ __forceinline__ float rmax16(float v) {
#pragma unroll
  for (int m = 1; m < 16; m <<= 1) v = fmaxf(v, __shfl_xor(v, m, 32));
  return v;
}
static __device__ __forceinline__ float rsum16(float v) {
#pragma unroll
  for (int m = 1; m < 16; m <<= 1) v += __shfl_xor(v, m, 32);
  return v;
}

// ---------------------------------------------------------------------------
// Embedding + sinusoidal positional encoding.  One block per token.
// ---------------------------------------------------------------------------
__global__ __launch_bounds__(256) void embed_kernel(
    const int* __restrict__ idx, const float* __restrict__ emb,
    float* __restrict__ out, int S) {
  const int token = blockIdx.x;
  const int s = token % S;
  const int id = idx[token];
  const float scale = 32.0f;                       // sqrt(1024)
  const float c = -logf(10000.0f) / (float)DMODEL;
#pragma unroll
  for (int i = 0; i < DMODEL / 256; ++i) {
    int j = threadIdx.x + i * 256;
    int jj = j & ~1;
    float ang = (float)s * __expf((float)jj * c);
    float pe = (j & 1) ? __cosf(ang) : __sinf(ang);
    out[token * DMODEL + j] = emb[id * DMODEL + j] * scale + pe;
  }
}

// ---------------------------------------------------------------------------
// Tiled GEMM: C[M,N] = act(A[M,K] x W[K,N] + bias) (+ res).
// Block = 8 waves, tile 64(M) x 128(N), K stepped by 32 through LDS (bf16).
// Each wave holds one B fragment and runs 4 WMMAs (4 M-subtiles).
// ---------------------------------------------------------------------------
template <bool RELU, bool HAS_RES>
__global__ __launch_bounds__(256) void gemm_kernel(
    const float* __restrict__ A, const float* __restrict__ W,
    const float* __restrict__ bias, const float* __restrict__ res,
    float* __restrict__ C, int M, int N, int K) {
  __shared__ unsigned short lA[64 * 32];
  __shared__ unsigned short lW[32 * 128];
  unsigned* lA32 = (unsigned*)lA;
  unsigned* lW32 = (unsigned*)lW;
  const int tid = threadIdx.x;
  const int w = tid >> 5, l = tid & 31, lo = l & 15, hi = l >> 4;
  const int m0 = blockIdx.y * 64;
  const int n0 = blockIdx.x * 128;
  v8f acc[4] = {{}, {}, {}, {}};
  for (int k0 = 0; k0 < K; k0 += 32) {
    __syncthreads();
#pragma unroll
    for (int i = 0; i < 2; ++i) {                  // A tile 64x32, float4 loads
      int idx = (tid + i * 256) * 4;
      int r = idx >> 5, cc = idx & 31;
      const float4 av = *(const float4*)&A[(m0 + r) * K + k0 + cc];
      lA32[(r * 32 + cc) >> 1]       = pack2bf(av.x, av.y);
      lA32[((r * 32 + cc) >> 1) + 1] = pack2bf(av.z, av.w);
    }
#pragma unroll
    for (int i = 0; i < 4; ++i) {                  // W tile 32x128, float4 loads
      int idx = (tid + i * 256) * 4;
      int kk = idx >> 7, nn = idx & 127;
      const float4 wv = *(const float4*)&W[(k0 + kk) * N + n0 + nn];
      lW32[(kk * 128 + nn) >> 1]       = pack2bf(wv.x, wv.y);
      lW32[((kk * 128 + nn) >> 1) + 1] = pack2bf(wv.z, wv.w);
    }
    if (k0 + 32 < K) __builtin_prefetch(&W[(k0 + 32) * N + n0], 0, 1);
    __syncthreads();
    v16bf b;
#pragma unroll
    for (int e = 0; e < 16; ++e)
      b[e] = u2bf(lW[(hi * 16 + e) * 128 + w * 16 + lo]);  // B layout
#pragma unroll
    for (int mt = 0; mt < 4; ++mt) {
      v16bf a;
#pragma unroll
      for (int e = 0; e < 16; ++e) {
        int ka = hi * 16 + e;
        a[e] = u2bf(lA[(mt * 16 + lo) * 32 + ka]);         // A layout
      }
      acc[mt] = __builtin_amdgcn_wmma_f32_16x16x32_bf16(false, a, false, b,
                                                        (short)0, acc[mt], false, false);
    }
  }
#pragma unroll
  for (int mt = 0; mt < 4; ++mt)
#pragma unroll
    for (int r = 0; r < 8; ++r) {
      int row = m0 + mt * 16 + r + hi * 8;
      int col = n0 + w * 16 + lo;
      float v = acc[mt][r] + bias[col];
      if (RELU) v = fmaxf(v, 0.0f);
      if (HAS_RES) v += res[row * N + col];
      C[row * N + col] = v;
    }
}

// ---------------------------------------------------------------------------
// Flash attention: one wave per 16-query tile, online softmax over 32-key
// tiles, QK^T and P*V via bf16 WMMA.  4 waves / block, per-wave LDS slice.
// All per-fragment loops fully unrolled (no MOVREL VGPR indexing).
// ---------------------------------------------------------------------------
__global__ __launch_bounds__(128) void attn_kernel(
    const float* __restrict__ Q, const float* __restrict__ Kmat,
    const float* __restrict__ Vmat, float* __restrict__ O,
    int Sq, int Skv, int causal) {
  __shared__ unsigned short pl[4][16][32];
  const int w = threadIdx.x >> 5, l = threadIdx.x & 31;
  const int lo = l & 15, hi = l >> 4;
  const int gw = blockIdx.x * 4 + w;
  const int ntq = Sq >> 4;
  const int qt = gw % ntq;
  const int bh = gw / ntq;
  const int h = bh % NHEAD, b = bh / NHEAD;

  const int qbase = (b * Sq + qt * 16) * DINT + h * DHEAD;
  v16bf a0, a1;                                    // Q fragments, d 0..31/32..63
#pragma unroll
  for (int e = 0; e < 16; ++e) {
    int ka = hi * 16 + e;
    a0[e] = fbf(Q[qbase + lo * DINT + ka]);
    a1[e] = fbf(Q[qbase + lo * DINT + 32 + ka]);
  }
  v8f oa[4] = {{}, {}, {}, {}};
  float mrun[8], lrun[8];
#pragma unroll
  for (int r = 0; r < 8; ++r) { mrun[r] = -1e30f; lrun[r] = 0.0f; }

  const int nkt = causal ? (qt / 2 + 1) : (Skv >> 5);
  for (int kt = 0; kt < nkt; ++kt) {
    const int kb = kt * 32;
    const int kbase0 = (b * Skv + kb + lo) * DINT + h * DHEAD;
    const int kbase1 = (b * Skv + kb + 16 + lo) * DINT + h * DHEAD;
    v16bf bk;
    v8f s0 = {}, s1 = {};
#pragma unroll
    for (int e = 0; e < 16; ++e) bk[e] = fbf(Kmat[kbase0 + hi * 16 + e]);
    s0 = __builtin_amdgcn_wmma_f32_16x16x32_bf16(false, a0, false, bk, (short)0, s0, false, false);
#pragma unroll
    for (int e = 0; e < 16; ++e) bk[e] = fbf(Kmat[kbase0 + 32 + hi * 16 + e]);
    s0 = __builtin_amdgcn_wmma_f32_16x16x32_bf16(false, a1, false, bk, (short)0, s0, false, false);
#pragma unroll
    for (int e = 0; e < 16; ++e) bk[e] = fbf(Kmat[kbase1 + hi * 16 + e]);
    s1 = __builtin_amdgcn_wmma_f32_16x16x32_bf16(false, a0, false, bk, (short)0, s1, false, false);
#pragma unroll
    for (int e = 0; e < 16; ++e) bk[e] = fbf(Kmat[kbase1 + 32 + hi * 16 + e]);
    s1 = __builtin_amdgcn_wmma_f32_16x16x32_bf16(false, a1, false, bk, (short)0, s1, false, false);

#pragma unroll
    for (int r = 0; r < 8; ++r) {                  // online softmax per row
      int qrow = qt * 16 + r + hi * 8;
      float v0 = s0[r] * 0.125f, v1 = s1[r] * 0.125f;   // 1/sqrt(64)
      if (causal && (kb + lo) > qrow) v0 = -1e30f;
      if (causal && (kb + 16 + lo) > qrow) v1 = -1e30f;
      float tm = rmax16(fmaxf(v0, v1));
      float mnew = fmaxf(mrun[r], tm);
      float al = __expf(mrun[r] - mnew);
      float p0 = __expf(v0 - mnew), p1 = __expf(v1 - mnew);
      float rs = rsum16(p0 + p1);
      lrun[r] = lrun[r] * al + rs;
      mrun[r] = mnew;
      oa[0][r] *= al; oa[1][r] *= al; oa[2][r] *= al; oa[3][r] *= al;
      pl[w][r + hi * 8][lo] = f2bfu(p0);           // C-layout -> LDS
      pl[w][r + hi * 8][16 + lo] = f2bfu(p1);
    }
    asm volatile("s_wait_dscnt 0" ::: "memory");   // LDS write->read in-wave
    __builtin_amdgcn_wave_barrier();
    v16bf ap, bv;
#pragma unroll
    for (int e = 0; e < 16; ++e) {                 // P as A-fragment 16x32
      int ka = hi * 16 + e;
      ap[e] = u2bf(pl[w][lo][ka]);
    }
    const int vb = (b * Skv + kb + hi * 16) * DINT + h * DHEAD;
#pragma unroll
    for (int nt = 0; nt < 4; ++nt) {               // P(16x32) x V(32x64)
#pragma unroll
      for (int e = 0; e < 16; ++e)
        bv[e] = fbf(Vmat[vb + e * DINT + nt * 16 + lo]);
      oa[nt] = __builtin_amdgcn_wmma_f32_16x16x32_bf16(false, ap, false, bv,
                                                       (short)0, oa[nt], false, false);
    }
  }
#pragma unroll
  for (int nt = 0; nt < 4; ++nt)
#pragma unroll
    for (int r = 0; r < 8; ++r)
      O[(b * Sq + qt * 16 + r + hi * 8) * DINT + h * DHEAD + nt * 16 + lo] =
          oa[nt][r] / lrun[r];
}

// ---------------------------------------------------------------------------
// LayerNorm over last dim (1024).  One block per token.
// ---------------------------------------------------------------------------
__global__ __launch_bounds__(256) void ln_kernel(
    const float* __restrict__ X, const float* __restrict__ g,
    const float* __restrict__ bta, float* __restrict__ O) {
  __shared__ float sh[8];
  const int token = blockIdx.x, tid = threadIdx.x;
  const float* row = X + token * DMODEL;
  float v[4], s = 0.0f;
#pragma unroll
  for (int i = 0; i < 4; ++i) { v[i] = row[tid + i * 256]; s += v[i]; }
#pragma unroll
  for (int m = 1; m < 32; m <<= 1) s += __shfl_xor(s, m, 32);
  if ((tid & 31) == 0) sh[tid >> 5] = s;
  __syncthreads();
  float mean = 0.0f;
#pragma unroll
  for (int i = 0; i < 8; ++i) mean += sh[i];
  mean *= (1.0f / DMODEL);
  __syncthreads();
  float q = 0.0f;
#pragma unroll
  for (int i = 0; i < 4; ++i) { float d = v[i] - mean; q += d * d; }
#pragma unroll
  for (int m = 1; m < 32; m <<= 1) q += __shfl_xor(q, m, 32);
  if ((tid & 31) == 0) sh[tid >> 5] = q;
  __syncthreads();
  float var = 0.0f;
#pragma unroll
  for (int i = 0; i < 8; ++i) var += sh[i];
  var *= (1.0f / DMODEL);
  float rstd = rsqrtf(var + 1e-5f);
#pragma unroll
  for (int i = 0; i < 4; ++i) {
    int j = tid + i * 256;
    O[token * DMODEL + j] = (v[i] - mean) * rstd * g[j] + bta[j];
  }
}

// ---------------------------------------------------------------------------
// Fused logits + chunked log-softmax stats.  64x128 strip per block reduced
// to per-(token,chunk) (max, sumexp, target-logit); the 262MB logits tensor
// is never materialized.
// ---------------------------------------------------------------------------
__global__ __launch_bounds__(256) void logits_kernel(
    const float* __restrict__ A, const float* __restrict__ W,
    const float* __restrict__ bias, const int* __restrict__ tgt,
    float* __restrict__ mb, float* __restrict__ sb, float* __restrict__ tb) {
  __shared__ unsigned short lA[64 * 32];
  __shared__ unsigned short lW[32 * 128];
  __shared__ float rm_s[8][64], rs_s[8][64], rt_s[8][64];
  unsigned* lA32 = (unsigned*)lA;
  unsigned* lW32 = (unsigned*)lW;
  const int tid = threadIdx.x;
  const int w = tid >> 5, l = tid & 31, lo = l & 15, hi = l >> 4;
  const int m0 = blockIdx.y * 64;
  const int n0 = blockIdx.x * 128;
  v8f acc[4] = {{}, {}, {}, {}};
  for (int k0 = 0; k0 < DMODEL; k0 += 32) {
    __syncthreads();
#pragma unroll
    for (int i = 0; i < 2; ++i) {
      int idx = (tid + i * 256) * 4;
      int r = idx >> 5, cc = idx & 31;
      const float4 av = *(const float4*)&A[(m0 + r) * DMODEL + k0 + cc];
      lA32[(r * 32 + cc) >> 1]       = pack2bf(av.x, av.y);
      lA32[((r * 32 + cc) >> 1) + 1] = pack2bf(av.z, av.w);
    }
#pragma unroll
    for (int i = 0; i < 4; ++i) {
      int idx = (tid + i * 256) * 4;
      int kk = idx >> 7, nn = idx & 127;
      const float4 wv = *(const float4*)&W[(k0 + kk) * VOCAB + n0 + nn];
      lW32[(kk * 128 + nn) >> 1]       = pack2bf(wv.x, wv.y);
      lW32[((kk * 128 + nn) >> 1) + 1] = pack2bf(wv.z, wv.w);
    }
    if (k0 + 32 < DMODEL) __builtin_prefetch(&W[(k0 + 32) * VOCAB + n0], 0, 1);
    __syncthreads();
    v16bf b;
#pragma unroll
    for (int e = 0; e < 16; ++e)
      b[e] = u2bf(lW[(hi * 16 + e) * 128 + w * 16 + lo]);
#pragma unroll
    for (int mt = 0; mt < 4; ++mt) {
      v16bf a;
#pragma unroll
      for (int e = 0; e < 16; ++e) {
        int ka = hi * 16 + e;
        a[e] = u2bf(lA[(mt * 16 + lo) * 32 + ka]);
      }
      acc[mt] = __builtin_amdgcn_wmma_f32_16x16x32_bf16(false, a, false, b,
                                                        (short)0, acc[mt], false, false);
    }
  }
#pragma unroll
  for (int mt = 0; mt < 4; ++mt)
#pragma unroll
    for (int r = 0; r < 8; ++r) {
      int rr = mt * 16 + r + hi * 8;
      int token = m0 + rr;
      int col = n0 + w * 16 + lo;
      float v = acc[mt][r] + bias[col];
      float rmax = rmax16(v);
      float rs = rsum16(__expf(v - rmax));
      int t = tgt[token];
      float tv = rmax16((col == t) ? v : -1e30f);
      if (lo == 0) { rm_s[w][rr] = rmax; rs_s[w][rr] = rs; rt_s[w][rr] = tv; }
    }
  __syncthreads();
  if (tid < 64) {                                  // merge 8 waves' stats
    float M = -1e30f, T = -1e30f;
#pragma unroll
    for (int w2 = 0; w2 < 8; ++w2) {
      M = fmaxf(M, rm_s[w2][tid]);
      T = fmaxf(T, rt_s[w2][tid]);
    }
    float Z = 0.0f;
#pragma unroll
    for (int w2 = 0; w2 < 8; ++w2) Z += rs_s[w2][tid] * __expf(rm_s[w2][tid] - M);
    int token = m0 + tid;
    mb[token * NCHUNK + blockIdx.x] = M;
    sb[token * NCHUNK + blockIdx.x] = Z;
    tb[token * NCHUNK + blockIdx.x] = T;
  }
}

// ---------------------------------------------------------------------------
// Merge chunks -> per-token target log-prob.
// ---------------------------------------------------------------------------
__global__ __launch_bounds__(256) void lse_kernel(
    const float* __restrict__ mb, const float* __restrict__ sb,
    const float* __restrict__ tb, float* __restrict__ tlp) {
  int token = blockIdx.x * 256 + threadIdx.x;
  if (token >= NTOK) return;
  float M = -1e30f, T = -1e30f;
  for (int c = 0; c < NCHUNK; ++c) {
    M = fmaxf(M, mb[token * NCHUNK + c]);
    T = fmaxf(T, tb[token * NCHUNK + c]);
  }
  float Z = 0.0f;
  for (int c = 0; c < NCHUNK; ++c)
    Z += sb[token * NCHUNK + c] * __expf(mb[token * NCHUNK + c] - M);
  tlp[token] = T - (M + __logf(Z));
}

// ---------------------------------------------------------------------------
// Per-batch masked sums.  out[0..3] = tot, out[4..7] = n_words (as float).
// ---------------------------------------------------------------------------
__global__ __launch_bounds__(256) void final_kernel(
    const float* __restrict__ tlp, const int* __restrict__ tgt,
    float* __restrict__ out) {
  __shared__ float sf[256];
  __shared__ int si[256];
  const int b = blockIdx.x, tid = threadIdx.x;
  float tot = 0.0f;
  int n = 0;
  for (int s = tid; s < SEQ; s += 256) {
    int token = b * SEQ + s;
    int t = tgt[token];
    int valid = (t != 0) && (t != 1);
    tot += tlp[token] * (float)valid;
    n += valid;
  }
  sf[tid] = tot; si[tid] = n;
  __syncthreads();
  for (int st = 128; st > 0; st >>= 1) {
    if (tid < st) { sf[tid] += sf[tid + st]; si[tid] += si[tid + st]; }
    __syncthreads();
  }
  if (tid == 0) { out[b] = sf[0]; out[BATCH + b] = (float)si[0]; }
}

// ---------------------------------------------------------------------------
// Host side
// ---------------------------------------------------------------------------
struct Lin { const float *W, *b; };
struct Blk { Lin q, k, v, o, f1, f2; const float *g1, *b1, *g2, *b2; };

static Blk grab(void* const* d, int& i) {
  Blk bk;
  bk.q.W  = (const float*)d[i++]; bk.q.b  = (const float*)d[i++];
  bk.k.W  = (const float*)d[i++]; bk.k.b  = (const float*)d[i++];
  bk.v.W  = (const float*)d[i++]; bk.v.b  = (const float*)d[i++];
  bk.o.W  = (const float*)d[i++]; bk.o.b  = (const float*)d[i++];
  bk.f1.W = (const float*)d[i++]; bk.f1.b = (const float*)d[i++];
  bk.f2.W = (const float*)d[i++]; bk.f2.b = (const float*)d[i++];
  bk.g1 = (const float*)d[i++];   bk.b1 = (const float*)d[i++];
  bk.g2 = (const float*)d[i++];   bk.b2 = (const float*)d[i++];
  return bk;
}

struct Scratch { float *X, *Y, *Q, *K, *V, *ATT, *T, *A, *MID, *MB, *SB, *TB, *TLP; };

static void run_block(const Blk& bk, float* xq, const float* xkv, int causal,
                      const Scratch& s, hipStream_t st) {
  dim3 gD(DINT / 128, NTOK / 64);                  // 1024-wide GEMMs
  gemm_kernel<false, false><<<gD, 256, 0, st>>>(xq,  bk.q.W, bk.q.b, nullptr, s.Q, NTOK, DINT, DMODEL);
  gemm_kernel<false, false><<<gD, 256, 0, st>>>(xkv, bk.k.W, bk.k.b, nullptr, s.K, NTOK, DINT, DMODEL);
  gemm_kernel<false, false><<<gD, 256, 0, st>>>(xkv, bk.v.W, bk.v.b, nullptr, s.V, NTOK, DINT, DMODEL);
  attn_kernel<<<(BATCH * NHEAD * (SEQ / 16)) / 4, 128, 0, st>>>(s.Q, s.K, s.V, s.ATT, SEQ, SEQ, causal);
  gemm_kernel<false, true><<<gD, 256, 0, st>>>(s.ATT, bk.o.W, bk.o.b, xq, s.T, NTOK, DMODEL, DINT);
  ln_kernel<<<NTOK, 256, 0, st>>>(s.T, bk.g1, bk.b1, s.A);
  gemm_kernel<true, false><<<dim3(2 * DMODEL / 128, NTOK / 64), 256, 0, st>>>(
      s.A, bk.f1.W, bk.f1.b, nullptr, s.MID, NTOK, 2 * DMODEL, DMODEL);
  gemm_kernel<false, true><<<gD, 256, 0, st>>>(s.MID, bk.f2.W, bk.f2.b, s.A, s.T, NTOK, DMODEL, 2 * DMODEL);
  ln_kernel<<<NTOK, 256, 0, st>>>(s.T, bk.g2, bk.b2, xq);
}

extern "C" void kernel_launch(void* const* d_in, const int* in_sizes, int n_in,
                              void* d_out, int out_size, void* d_ws, size_t ws_size,
                              hipStream_t stream) {
  (void)in_sizes; (void)n_in; (void)out_size; (void)ws_size;
  const int* src = (const int*)d_in[0];
  const int* tgt = (const int*)d_in[1];
  int i = 2;
  const float* src_emb = (const float*)d_in[i++];
  const float* tgt_emb = (const float*)d_in[i++];
  Lin outp;
  outp.W = (const float*)d_in[i++];
  outp.b = (const float*)d_in[i++];
  Blk enc[NLAYER], dd[NLAYER], ed[NLAYER];
  for (int l = 0; l < NLAYER; ++l) enc[l] = grab(d_in, i);
  for (int l = 0; l < NLAYER; ++l) { dd[l] = grab(d_in, i); ed[l] = grab(d_in, i); }

  // Workspace layout (floats)
  float* ws = (float*)d_ws;
  const size_t TD = (size_t)NTOK * DMODEL;         // 2,097,152
  Scratch s;
  size_t off = 0;
  s.X   = ws + off; off += TD;
  s.Y   = ws + off; off += TD;
  s.Q   = ws + off; off += TD;
  s.K   = ws + off; off += TD;
  s.V   = ws + off; off += TD;
  s.ATT = ws + off; off += TD;
  s.T   = ws + off; off += TD;
  s.A   = ws + off; off += TD;
  s.MID = ws + off; off += (size_t)NTOK * 2 * DMODEL;
  s.MB  = ws + off; off += (size_t)NTOK * NCHUNK;
  s.SB  = ws + off; off += (size_t)NTOK * NCHUNK;
  s.TB  = ws + off; off += (size_t)NTOK * NCHUNK;
  s.TLP = ws + off; off += NTOK;

  // Encoder
  embed_kernel<<<NTOK, 256, 0, stream>>>(src, src_emb, s.X, SEQ);
  for (int l = 0; l < NLAYER; ++l) run_block(enc[l], s.X, s.X, 0, s, stream);

  // Decoder
  embed_kernel<<<NTOK, 256, 0, stream>>>(tgt, tgt_emb, s.Y, SEQ);
  for (int l = 0; l < NLAYER; ++l) {
    run_block(dd[l], s.Y, s.Y, 1, s, stream);      // masked self-attn
    run_block(ed[l], s.Y, s.X, 0, s, stream);      // cross-attn vs encoder out
  }

  // Fused logits -> chunked log-softmax -> per-batch sums
  logits_kernel<<<dim3(NCHUNK, NTOK / 64), 256, 0, stream>>>(
      s.Y, outp.W, outp.b, tgt, s.MB, s.SB, s.TB);
  lse_kernel<<<NTOK / 256, 256, 0, stream>>>(s.MB, s.SB, s.TB, s.TLP);
  final_kernel<<<BATCH, 256, 0, stream>>>(s.TLP, tgt, (float*)d_out);
}
